// Head_51762945852019
// MI455X (gfx1250) — matrix operand order repaired and checked by reading, not compiled, not run on hardware
//
#include <hip/hip_runtime.h>

typedef __attribute__((ext_vector_type(16))) _Float16 v16h;
typedef __attribute__((ext_vector_type(8)))  _Float16 h8;
typedef __attribute__((ext_vector_type(8)))  float    v8f;
typedef __attribute__((ext_vector_type(4)))  _Float16 h4;

#define T_SEQ 64
#define C_DIM 48
#define H_DIM 48
#define LDA   72   // LDS leading dim: 144B row stride (36-bank stride, conflict-free),
                   // multiple of 8 halves so all b128 fragment accesses stay 16B-aligned
#define BPB   4    // batches per block: amortizes weight staging + pad zeroing

// ---------------- LDS layout (single buffer, halves) ----------------
#define OFF_X   0                       // X   [64][LDA]  t x c  (c padded 48->64)
#define OFF_WQ  (OFF_X  + T_SEQ * LDA)  // WqT [48][LDA]  h x c  (transposed, c padded)
#define OFF_WK  (OFF_WQ + H_DIM * LDA)
#define OFF_WV  (OFF_WK + H_DIM * LDA)
#define OFF_Q   (OFF_WV + H_DIM * LDA)  // Q   [64][LDA]  t x h  (h padded)
#define OFF_K   (OFF_Q  + T_SEQ * LDA)  // K   [64][LDA]  s x h  (h padded)
#define OFF_V   (OFF_K  + T_SEQ * LDA)  // VT  [48][LDA]  h x s  (s = 64, no pad)
#define OFF_P   (OFF_V  + H_DIM * LDA)  // P   [64][LDA]  t x s  (s = 64, no pad)
#define SMEM_HALVES (OFF_P + T_SEQ * LDA)

// ---- fragment loaders: unconditional, 2 x ds_load_b128 each (§7.12.2 layouts) ----
__device__ __forceinline__ v16h frag_A(const _Float16* src, int m0, int k0,
                                       int lo, int hi) {
  const _Float16* p = src + (m0 + lo) * LDA + k0 + hi * 8;
  const h8 a0 = *(const h8*)(p);
  const h8 a1 = *(const h8*)(p + 16);
  return __builtin_shufflevector(a0, a1, 0, 1, 2, 3, 4, 5, 6, 7,
                                 8, 9, 10, 11, 12, 13, 14, 15);
}
__device__ __forceinline__ v16h frag_B(const _Float16* src, int n0, int k0,
                                       int lo, int hi) {
  const _Float16* p = src + (n0 + lo) * LDA + k0 + hi * 8;
  const h8 b0 = *(const h8*)(p);
  const h8 b1 = *(const h8*)(p + 16);
  return __builtin_shufflevector(b0, b1, 0, 1, 2, 3, 4, 5, 6, 7,
                                 8, 9, 10, 11, 12, 13, 14, 15);
}

#define WMMA_F16(A, B, Cc) \
  __builtin_amdgcn_wmma_f32_16x16x32_f16(false, (A), false, (B), (short)0, (Cc), false, false)

__device__ __forceinline__ void zero_pad16(_Float16* p) {
  const h8 z = {};
  *(h8*)(p)     = z;
  *(h8*)(p + 8) = z;
}

__global__ __launch_bounds__(128, 1) void attn_head_fused_kernel(
    const float* __restrict__ x, const float* __restrict__ Wq,
    const float* __restrict__ Wk, const float* __restrict__ Wv,
    float* __restrict__ out, int n_batch) {
  __shared__ __align__(16) _Float16 smem[SMEM_HALVES];
  _Float16* Xh  = smem + OFF_X;
  _Float16* WqT = smem + OFF_WQ;
  _Float16* WkT = smem + OFF_WK;
  _Float16* WvT = smem + OFF_WV;
  _Float16* Qh  = smem + OFF_Q;
  _Float16* Kh  = smem + OFF_K;
  _Float16* VT  = smem + OFF_V;
  _Float16* Pst = smem + OFF_P;

  const int tid   = threadIdx.x;
  const int lane  = tid & 31;
  const int wave  = tid >> 5;
  const int lo    = lane & 15;
  const int hi    = lane >> 4;
  const int tile0 = wave * 16;  // this wave's 16-wide t-tile (and s-strip base)

  // ---- one-time setup: zero K-pad columns (48..63); never overwritten later ----
  if (tid < T_SEQ) {
    zero_pad16(&Xh[tid * LDA + C_DIM]);
    zero_pad16(&Qh[tid * LDA + H_DIM]);
    zero_pad16(&Kh[tid * LDA + H_DIM]);
  }
  if (tid < H_DIM) {
    zero_pad16(&WqT[tid * LDA + C_DIM]);
    zero_pad16(&WkT[tid * LDA + C_DIM]);
    zero_pad16(&WvT[tid * LDA + C_DIM]);
  }
  // ---- one-time setup: stage weights TRANSPOSED (h-major, c-contiguous) ----
#pragma unroll
  for (int i = tid * 2; i < C_DIM * H_DIM; i += 128 * 2) {
    const int c = i / H_DIM;
    const int h = i % H_DIM;
    const float2 q = *(const float2*)(Wq + i);
    const float2 k = *(const float2*)(Wk + i);
    const float2 v = *(const float2*)(Wv + i);
    WqT[(h)     * LDA + c] = (_Float16)q.x;
    WqT[(h + 1) * LDA + c] = (_Float16)q.y;
    WkT[(h)     * LDA + c] = (_Float16)k.x;
    WkT[(h + 1) * LDA + c] = (_Float16)k.y;
    WvT[(h)     * LDA + c] = (_Float16)v.x;
    WvT[(h + 1) * LDA + c] = (_Float16)v.y;
  }

  for (int bb = 0; bb < BPB; ++bb) {
    const int b = blockIdx.x * BPB + bb;
    if (b >= n_batch) break;

    // ---- stage x batch tile (fp32 -> f16), float4-vectorized ----
    // Only touches Xh: safe to overlap with the previous batch's phase 3.
    const float* xb = x + (size_t)b * (T_SEQ * C_DIM);
#pragma unroll
    for (int i = tid * 4; i < T_SEQ * C_DIM; i += 128 * 4) {
      const int t = i / C_DIM;         // 48 % 4 == 0: chunk never crosses a row
      const int c = i % C_DIM;
      const float4 f = *(const float4*)(xb + i);
      h4 h;
      h[0] = (_Float16)f.x; h[1] = (_Float16)f.y;
      h[2] = (_Float16)f.z; h[3] = (_Float16)f.w;
      *(h4*)(&Xh[t * LDA + c]) = h;
    }
    __syncthreads();  // staging done; also separates prev phase-3 reads from phase-1 writes

    // ==== Phase 1: Q^T = Wq^T @ X^T, K^T = Wk^T @ X^T, V = X @ Wv ====
    // D tiles land with each lane's 8 registers contiguous in the dest row
    // -> one packed b128 LDS store per tile.
    const v16h xb0 = frag_B(Xh, tile0, 0, lo, hi);
    const v16h xb1 = frag_B(Xh, tile0, 32, lo, hi);
    const v16h xa0 = frag_A(Xh, tile0, 0, lo, hi);
    const v16h xa1 = frag_A(Xh, tile0, 32, lo, hi);
#pragma unroll
    for (int ht = 0; ht < 3; ++ht) {
      const int h0 = ht * 16;
      v8f aq = {}, ak = {}, av = {};
      aq = WMMA_F16(frag_A(WqT, h0, 0, lo, hi), xb0, aq);   // D[m=h][n=t]
      aq = WMMA_F16(frag_A(WqT, h0, 32, lo, hi), xb1, aq);
      ak = WMMA_F16(frag_A(WkT, h0, 0, lo, hi), xb0, ak);   // D[m=h][n=s]
      ak = WMMA_F16(frag_A(WkT, h0, 32, lo, hi), xb1, ak);
      av = WMMA_F16(xa0, frag_B(WvT, h0, 0, lo, hi), av);   // D[m=s][n=h]
      av = WMMA_F16(xa1, frag_B(WvT, h0, 32, lo, hi), av);
      h8 hq, hk, hv;
#pragma unroll
      for (int r = 0; r < 8; ++r) {
        hq[r] = (_Float16)aq[r];
        hk[r] = (_Float16)ak[r];
        hv[r] = (_Float16)av[r];
      }
      *(h8*)(&Qh[(tile0 + lo) * LDA + h0 + 8 * hi]) = hq;   // Q[t][h]
      *(h8*)(&Kh[(tile0 + lo) * LDA + h0 + 8 * hi]) = hk;   // K[s][h]
      *(h8*)(&VT[(h0 + lo) * LDA + tile0 + 8 * hi]) = hv;   // VT[h][s]
    }
    __syncthreads();

    // ==== Phase 2: S^T = K @ Q^T (lane owns one t; 8 contiguous s per tile) ====
    const v16h qb0 = frag_B(Qh, tile0, 0, lo, hi);
    const v16h qb1 = frag_B(Qh, tile0, 32, lo, hi);
    const int t = tile0 + lo;
    float sc[4][8];
#pragma unroll
    for (int st = 0; st < 4; ++st) {
      const int s0 = st * 16;
      v8f acc = {};
      acc = WMMA_F16(frag_A(Kh, s0, 0, lo, hi), qb0, acc);  // D[m=s][n=t]
      acc = WMMA_F16(frag_A(Kh, s0, 32, lo, hi), qb1, acc);
#pragma unroll
      for (int r = 0; r < 8; ++r) sc[st][r] = acc[r];
    }
    // scale folded with log2(e): softmax computed in exp2 domain (v_exp_f32 is exp2)
    const float c2 = 0.14433756729740643f * 1.4426950408889634f;  // 1/sqrt(48) * log2(e)
    float mx = -1.0e30f;
#pragma unroll
    for (int st = 0; st < 4; ++st) {
#pragma unroll
      for (int r = 0; r < 8; ++r) {
        const int s = st * 16 + r + 8 * hi;
        sc[st][r] = (s <= t) ? sc[st][r] * c2 : -1.0e30f;
        mx = fmaxf(mx, sc[st][r]);
      }
    }
    // Partner half-lane (lane^16) holds the other 32 s-values of the same t.
    mx = fmaxf(mx, __shfl_xor(mx, 16, 32));
    float sum = 0.0f;
#pragma unroll
    for (int st = 0; st < 4; ++st)
#pragma unroll
      for (int r = 0; r < 8; ++r) {
        sc[st][r] = __builtin_amdgcn_exp2f(sc[st][r] - mx);  // v_exp_f32 (hw exp2)
        sum += sc[st][r];
      }
    sum += __shfl_xor(sum, 16, 32);
    const float inv = 1.0f / sum;
#pragma unroll
    for (int st = 0; st < 4; ++st) {
      h8 hp;
#pragma unroll
      for (int r = 0; r < 8; ++r) hp[r] = (_Float16)(sc[st][r] * inv);
      *(h8*)(&Pst[t * LDA + st * 16 + 8 * hi]) = hp;        // P[t][s], s-contiguous
    }
    __syncthreads();

    // ==== Phase 3: O^T = V^T @ P^T ; lane owns one t, 8 contiguous h per tile ====
    const v16h pb0 = frag_B(Pst, tile0, 0, lo, hi);
    const v16h pb1 = frag_B(Pst, tile0, 32, lo, hi);
    float* ob = out + (size_t)b * (T_SEQ * H_DIM) + (size_t)t * H_DIM;
#pragma unroll
    for (int ht = 0; ht < 3; ++ht) {
      const int h0 = ht * 16;
      v8f acc = {};
      acc = WMMA_F16(frag_A(VT, h0, 0, lo, hi), pb0, acc);  // D[m=h][n=t]
      acc = WMMA_F16(frag_A(VT, h0, 32, lo, hi), pb1, acc);
      float* dst = ob + h0 + 8 * hi;   // 32B-aligned, 8 contiguous floats
      *(float4*)(dst)     = make_float4(acc[0], acc[1], acc[2], acc[3]);
      *(float4*)(dst + 4) = make_float4(acc[4], acc[5], acc[6], acc[7]);
    }
    // no barrier here: next iteration's X staging doesn't touch VT/Pst, and the
    // staging barrier orders phase-3 reads vs. the next phase-1 LDS writes.
  }
}

extern "C" void kernel_launch(void* const* d_in, const int* in_sizes, int n_in,
                              void* d_out, int out_size, void* d_ws, size_t ws_size,
                              hipStream_t stream) {
  const float* x  = (const float*)d_in[0];
  const float* Wq = (const float*)d_in[1];
  const float* Wk = (const float*)d_in[2];
  const float* Wv = (const float*)d_in[3];
  float* out = (float*)d_out;
  const int B = in_sizes[0] / (T_SEQ * C_DIM);  // 16384
  const int grid = (B + BPB - 1) / BPB;
  attn_head_fused_kernel<<<grid, 128, 0, stream>>>(x, Wq, Wk, Wv, out, B);
}
